// Cross_Attention_9766755631565
// MI455X (gfx1250) — matrix operand order repaired and checked by reading, not compiled, not run on hardware
//
#include <hip/hip_runtime.h>
#include <hip/hip_bf16.h>

// ---------------------------------------------------------------------------
// Cross-attention cost-volume block for MI455X (gfx1250, wave32, WMMA).
//
// Shapes: Cost (1,32,48,64,128) f32, Fmap (1,96,64,128) f32.
//   P = 48*64*128 = 393216 positions, S = 64*128 = 8192.
// Heavy stages (5x5 conv and all 1x1 convs) run as bf16 implicit GEMMs on
// v_wmma_f32_16x16x32_bf16; small L2-resident stages run on VALU.
// ---------------------------------------------------------------------------

typedef __bf16 v16bf __attribute__((ext_vector_type(16)));
typedef float v8f   __attribute__((ext_vector_type(8)));

#define CC   32
#define DD   48
#define HH   64
#define WWID 128
#define PP   (DD * HH * WWID)   // 393216
#define SS   (HH * WWID)        // 8192
#define EPSF 1e-5f

// K-index of element e in a bf16 A-fragment (ISA 7.12.2, 16-bit A 16x32):
//   lanes 0-15 : K = {0..7,16..23}; lanes 16-31: +8
__device__ __forceinline__ int kkA(int e, int half) {
    return e + ((e >> 3) << 3) + half * 8;
}

// ---------------------------------------------------------------------------
// Weight pre-swizzle: pack fp32 weights into bf16 B-fragments
// (B 32x16: lanes 0-15 hold K={0..7,16..23} of column N=lane; lanes 16-31: +8).
// ---------------------------------------------------------------------------

// qk_w1 (32 cout, 32 cin, 1, 5, 5) -> wf5[tap][frag][lane][16]
__global__ __launch_bounds__(64) void k_prep_w5(const float* __restrict__ w,
                                                __bf16* __restrict__ out) {
    const int t = blockIdx.x;            // tap 0..24
    const int f = threadIdx.x >> 5;      // frag 0/1 (cout 0-15 / 16-31)
    const int l = threadIdx.x & 31;
    const int half = l >> 4, lr = l & 15;
    const int cout = f * 16 + lr;
    #pragma unroll
    for (int e = 0; e < 16; ++e) {
        const int cin = kkA(e, half);
        out[((t * 2 + f) * 32 + l) * 16 + e] = (__bf16)w[(cout * 32 + cin) * 25 + t];
    }
}

// 32x32 pointwise weight (cout,cin) -> wf[frag][lane][16]
__global__ __launch_bounds__(64) void k_prep_pw(const float* __restrict__ w,
                                                __bf16* __restrict__ out) {
    const int f = threadIdx.x >> 5;
    const int l = threadIdx.x & 31;
    const int half = l >> 4, lr = l & 15;
    const int cout = f * 16 + lr;
    #pragma unroll
    for (int e = 0; e < 16; ++e) {
        const int cin = kkA(e, half);
        out[(f * 32 + l) * 16 + e] = (__bf16)w[cout * 32 + cin];
    }
}

// ---------------------------------------------------------------------------
// Kernel 1: qk conv3d (1,5,5) 32->32 + BN + leaky ReLU, bf16 WMMA.
// grid = D*H*(W/64); block = 128 (4 waves); each wave: 16 w-positions x 32 cout.
// ---------------------------------------------------------------------------
__global__ __launch_bounds__(128) void k_conv5x5(
    const float* __restrict__ x,            // [32][P]
    const v16bf* __restrict__ wfrag,        // [25][2][32] fragments
    const float* __restrict__ bg, const float* __restrict__ bb,
    const float* __restrict__ bm, const float* __restrict__ bv,
    float* __restrict__ y)                  // [32][P]
{
    __shared__ __bf16 lin[5][68][32];       // [row][w(+halo)][cin], 21.25 KB
    __shared__ float sscale[32], sshift[32];

    const int tid = threadIdx.x;
    const int b   = blockIdx.x;
    const int wt  = b & 1;
    const int h   = (b >> 1) & (HH - 1);
    const int d   = b >> 7;
    const int w0  = wt * 64;

    if (tid < 32) {
        const float sc = bg[tid] * rsqrtf(bv[tid] + EPSF);
        sscale[tid] = sc;
        sshift[tid] = bb[tid] - bm[tid] * sc;
    }
    // Stage input tile (rows h-2..h+2, w w0-2..w0+65, all 32 cin) as bf16.
    // wl is the fastest index so consecutive lanes read consecutive global
    // addresses along W (coalesced HBM segments on the dominant 50MB read).
    for (int i = tid; i < 5 * 32 * 68; i += 128) {
        const int wl  = i % 68;
        const int cin = (i / 68) & 31;
        const int r   = i / (68 * 32);
        const int gh  = h + r - 2;
        const int gw  = w0 + wl - 2;
        float val = 0.f;
        if ((unsigned)gh < (unsigned)HH && (unsigned)gw < (unsigned)WWID)
            val = x[cin * PP + (d * HH + gh) * WWID + gw];
        lin[r][wl][cin] = (__bf16)val;
    }
    __syncthreads();

    const int wave = tid >> 5, lane = tid & 31;
    const int half = lane >> 4, lrow = lane & 15;

    v8f acc0 = {}; v8f acc1 = {};
    #pragma unroll 1
    for (int t = 0; t < 25; ++t) {
        const int dy = t / 5, dx = t % 5;
        const int p  = wave * 16 + lrow + dx;
        // A fragment: 8 dword LDS reads (two runs of 4) in the ISA lane layout
        const unsigned int* row32 = (const unsigned int*)(&lin[dy][p][0]);
        union { unsigned int u[8]; v16bf v; } au;
        #pragma unroll
        for (int j = 0; j < 4; ++j) {
            au.u[j]     = row32[half * 4 + j];
            au.u[4 + j] = row32[8 + half * 4 + j];
        }
        const v16bf bA = wfrag[(t * 2 + 0) * 32 + lane];
        const v16bf bB = wfrag[(t * 2 + 1) * 32 + lane];
        acc0 = __builtin_amdgcn_wmma_f32_16x16x32_bf16(false, au.v, false, bA,
                                                       (short)0, acc0, false, false);
        acc1 = __builtin_amdgcn_wmma_f32_16x16x32_bf16(false, au.v, false, bB,
                                                       (short)0, acc1, false, false);
    }

    // epilogue: D layout lane N=cout (l&15), VGPR r -> M = r + (l>>4)*8
    const float sc0 = sscale[lrow],      sh0 = sshift[lrow];
    const float sc1 = sscale[lrow + 16], sh1 = sshift[lrow + 16];
    const int pbase = (d * HH + h) * WWID + w0 + wave * 16;
    #pragma unroll
    for (int r = 0; r < 8; ++r) {
        const int m = r + half * 8;
        float y0 = acc0[r] * sc0 + sh0; y0 = (y0 > 0.f) ? y0 : 0.01f * y0;
        float y1 = acc1[r] * sc1 + sh1; y1 = (y1 > 0.f) ? y1 : 0.01f * y1;
        y[lrow * PP + pbase + m]        = y0;
        y[(lrow + 16) * PP + pbase + m] = y1;
    }
}

// ---------------------------------------------------------------------------
// Kernel 2: generic 32->32 pointwise GEMM over P positions (bf16 WMMA).
// mode 0: BN + leaky  (p0..p3 = g,b,m,v)
// mode 1: + bias      (p0 = bias)
// mode 2: sigmoid(.+bias) * other
// grid = P / 1024; block = 256 (8 waves x 8 tiles x 16 positions)
// ---------------------------------------------------------------------------
__global__ __launch_bounds__(256) void k_pointwise32(
    const float* __restrict__ x, const v16bf* __restrict__ wf,
    const float* __restrict__ p0, const float* __restrict__ p1,
    const float* __restrict__ p2, const float* __restrict__ p3,
    const float* __restrict__ other, float* __restrict__ y, int mode)
{
    __shared__ float sA[32], sB[32];
    const int tid = threadIdx.x;
    if (tid < 32) {
        if (mode == 0) {
            const float sc = p0[tid] * rsqrtf(p3[tid] + EPSF);
            sA[tid] = sc; sB[tid] = p1[tid] - p2[tid] * sc;
        } else {
            sA[tid] = 1.f; sB[tid] = p0[tid];
        }
    }
    __syncthreads();

    const int lane = tid & 31, wave = tid >> 5;
    const int half = lane >> 4, lrow = lane & 15;
    const v16bf bA = wf[lane];
    const v16bf bB = wf[32 + lane];
    const float c0s = sA[lrow],      c0b = sB[lrow];
    const float c1s = sA[lrow + 16], c1b = sB[lrow + 16];
    const int pwave = (blockIdx.x * 8 + wave) * 128;

    #pragma unroll 1
    for (int it = 0; it < 8; ++it) {
        const int pbase = pwave + it * 16;
        const int pos   = pbase + lrow;
        v16bf a;
        #pragma unroll
        for (int e = 0; e < 16; ++e)
            a[e] = (__bf16)x[kkA(e, half) * PP + pos];  // coalesced across lanes per e
        v8f acc0 = {}; v8f acc1 = {};
        acc0 = __builtin_amdgcn_wmma_f32_16x16x32_bf16(false, a, false, bA,
                                                       (short)0, acc0, false, false);
        acc1 = __builtin_amdgcn_wmma_f32_16x16x32_bf16(false, a, false, bB,
                                                       (short)0, acc1, false, false);
        #pragma unroll
        for (int r = 0; r < 8; ++r) {
            const int pp = pbase + r + half * 8;
            float y0 = acc0[r] * c0s + c0b;
            float y1 = acc1[r] * c1s + c1b;
            if (mode == 0) {
                y0 = (y0 > 0.f) ? y0 : 0.01f * y0;
                y1 = (y1 > 0.f) ? y1 : 0.01f * y1;
            } else if (mode == 2) {
                y0 = other[lrow * PP + pp]        * (1.f / (1.f + __expf(-y0)));
                y1 = other[(lrow + 16) * PP + pp] * (1.f / (1.f + __expf(-y1)));
            }
            y[lrow * PP + pp]        = y0;
            y[(lrow + 16) * PP + pp] = y1;
        }
    }
}

// ---------------------------------------------------------------------------
// Kernel 3: windowed attention logits + softmax over K=9. qk stays in L2.
// att layout: [m*9+k][P]
// ---------------------------------------------------------------------------
__global__ __launch_bounds__(256) void k_att(const float* __restrict__ qk,
                                             float* __restrict__ att)
{
    const int p = blockIdx.x * 256 + threadIdx.x;
    const int wpos = p & (WWID - 1);
    const int hpos = (p >> 7) & (HH - 1);
    float q[32];
    #pragma unroll
    for (int ch = 0; ch < 32; ++ch) q[ch] = qk[ch * PP + p];
    #pragma unroll 1
    for (int m = 0; m < 4; ++m) {
        float lg[9];
        #pragma unroll
        for (int k = 0; k < 9; ++k) {
            const int dy = k / 3 - 1, dx = k % 3 - 1;
            const int nh = hpos + dy, nw = wpos + dx;
            float s = 0.f;
            if ((unsigned)nh < (unsigned)HH && (unsigned)nw < (unsigned)WWID) {
                const int np = p + dy * WWID + dx;
                #pragma unroll
                for (int cc = 0; cc < 8; ++cc)
                    s += q[m * 8 + cc] * qk[(m * 8 + cc) * PP + np];
            }
            lg[k] = s;   // zero-padded window -> logit 0 participates in softmax
        }
        float mx = lg[0];
        #pragma unroll
        for (int k = 1; k < 9; ++k) mx = fmaxf(mx, lg[k]);
        float sum = 0.f;
        #pragma unroll
        for (int k = 0; k < 9; ++k) { lg[k] = __expf(lg[k] - mx); sum += lg[k]; }
        const float inv = 1.f / sum;
        #pragma unroll
        for (int k = 0; k < 9; ++k) att[(m * 9 + k) * PP + p] = lg[k] * inv;
    }
}

// ---------------------------------------------------------------------------
// to_v path (tiny, L2-resident): conv2d 3x3 96->96 + BN + leaky
// ---------------------------------------------------------------------------
__global__ __launch_bounds__(256) void k_conv3x3_v(
    const float* __restrict__ x, const float* __restrict__ w,
    const float* __restrict__ bg, const float* __restrict__ bb,
    const float* __restrict__ bm, const float* __restrict__ bv,
    float* __restrict__ y)
{
    __shared__ float lw[96 * 9];
    const int cout = blockIdx.x >> 5;
    const int s0   = (blockIdx.x & 31) * 256;
    for (int i = threadIdx.x; i < 96 * 9; i += 256) lw[i] = w[cout * 96 * 9 + i];
    __syncthreads();
    const int s = s0 + threadIdx.x;
    const int hpos = s >> 7, wpos = s & (WWID - 1);
    float acc = 0.f;
    #pragma unroll 1
    for (int ci = 0; ci < 96; ++ci) {
        const float* xc = x + ci * SS;
        const float* wc = &lw[ci * 9];
        #pragma unroll
        for (int t = 0; t < 9; ++t) {
            const int nh = hpos + t / 3 - 1, nw = wpos + t % 3 - 1;
            if ((unsigned)nh < (unsigned)HH && (unsigned)nw < (unsigned)WWID)
                acc += wc[t] * xc[nh * WWID + nw];
        }
    }
    const float sc = bg[cout] * rsqrtf(bv[cout] + EPSF);
    acc = (acc - bm[cout]) * sc + bb[cout];
    y[cout * SS + s] = (acc > 0.f) ? acc : 0.01f * acc;
}

// small pointwise over S positions; mode 0: BN+leaky, mode 1: +bias
__global__ __launch_bounds__(256) void k_pw_small(
    const float* __restrict__ x, const float* __restrict__ w,
    int CIN, int COUT,
    const float* __restrict__ q0, const float* __restrict__ q1,
    const float* __restrict__ q2, const float* __restrict__ q3,
    float* __restrict__ y, int mode)
{
    const int idx = blockIdx.x * 256 + threadIdx.x;
    if (idx >= COUT * SS) return;
    const int cout = idx >> 13;
    const int s    = idx & (SS - 1);
    float acc = 0.f;
    for (int ci = 0; ci < CIN; ++ci) acc += w[cout * CIN + ci] * x[ci * SS + s];
    if (mode == 0) {
        const float sc = q0[cout] * rsqrtf(q3[cout] + EPSF);
        acc = (acc - q2[cout]) * sc + q1[cout];
        acc = (acc > 0.f) ? acc : 0.01f * acc;
    } else {
        acc += q0[cout];
    }
    y[idx] = acc;
}

// ---------------------------------------------------------------------------
// excite = einsum(att, unfold(v)) + channel LayerNorm
// ---------------------------------------------------------------------------
__global__ __launch_bounds__(256) void k_excite_ln(
    const float* __restrict__ att, const float* __restrict__ v,
    const float* __restrict__ lng, const float* __restrict__ lnb,
    float* __restrict__ y)
{
    const int p = blockIdx.x * 256 + threadIdx.x;
    const int wpos = p & (WWID - 1);
    const int hpos = (p >> 7) & (HH - 1);
    float ex[32];
    #pragma unroll
    for (int ch = 0; ch < 32; ++ch) ex[ch] = 0.f;
    #pragma unroll 1
    for (int k = 0; k < 9; ++k) {
        const int nh = hpos + k / 3 - 1, nw = wpos + k % 3 - 1;
        if ((unsigned)nh >= (unsigned)HH || (unsigned)nw >= (unsigned)WWID) continue;
        const int sn = nh * WWID + nw;
        const float a0 = att[(0 * 9 + k) * PP + p];
        const float a1 = att[(1 * 9 + k) * PP + p];
        const float a2 = att[(2 * 9 + k) * PP + p];
        const float a3 = att[(3 * 9 + k) * PP + p];
        #pragma unroll
        for (int cc = 0; cc < 8; ++cc) {
            ex[cc]      += a0 * v[cc * SS + sn];
            ex[8 + cc]  += a1 * v[(8 + cc) * SS + sn];
            ex[16 + cc] += a2 * v[(16 + cc) * SS + sn];
            ex[24 + cc] += a3 * v[(24 + cc) * SS + sn];
        }
    }
    float mu = 0.f;
    #pragma unroll
    for (int ch = 0; ch < 32; ++ch) mu += ex[ch];
    mu *= (1.f / 32.f);
    float var = 0.f;
    #pragma unroll
    for (int ch = 0; ch < 32; ++ch) { const float t = ex[ch] - mu; var += t * t; }
    var *= (1.f / 32.f);
    const float inv = rsqrtf(var + EPSF);
    #pragma unroll
    for (int ch = 0; ch < 32; ++ch)
        y[ch * PP + p] = (ex[ch] - mu) * inv * lng[ch] + lnb[ch];
}

// ---------------------------------------------------------------------------
// Orchestration. Input order follows setup_inputs() dict flattening:
//  0 Cost, 1 Fmap, 2 qk_w1, 3-6 qk_bn(g,b,m,v), 7 qk_w2, 8 qk_b2,
//  9 v_w1, 10-13 v_bn1, 14 v_w2, 15-18 v_bn2, 19 v_w3, 20 v_b3,
//  21 a_w1, 22-25 a_bn, 26 a_w2, 27 a_b2, 28 g_w1, 29-32 g_bn,
//  33 g_w2, 34 g_b2, 35 ln_g, 36 ln_b
// Workspace: ~164 MB (bufA 32P, bufB 32P, bufC 36P, v bufs, bf16 weight frags)
// ---------------------------------------------------------------------------
extern "C" void kernel_launch(void* const* d_in, const int* in_sizes, int n_in,
                              void* d_out, int out_size, void* d_ws, size_t ws_size,
                              hipStream_t stream) {
    const float* Cost = (const float*)d_in[0];
    const float* Fmap = (const float*)d_in[1];
    const float* qk_w1 = (const float*)d_in[2];
    const float* qk_bg = (const float*)d_in[3];
    const float* qk_bb = (const float*)d_in[4];
    const float* qk_bm = (const float*)d_in[5];
    const float* qk_bv = (const float*)d_in[6];
    const float* qk_w2 = (const float*)d_in[7];
    const float* qk_b2 = (const float*)d_in[8];
    const float* v_w1  = (const float*)d_in[9];
    const float* v1_bg = (const float*)d_in[10];
    const float* v1_bb = (const float*)d_in[11];
    const float* v1_bm = (const float*)d_in[12];
    const float* v1_bv = (const float*)d_in[13];
    const float* v_w2  = (const float*)d_in[14];
    const float* v2_bg = (const float*)d_in[15];
    const float* v2_bb = (const float*)d_in[16];
    const float* v2_bm = (const float*)d_in[17];
    const float* v2_bv = (const float*)d_in[18];
    const float* v_w3  = (const float*)d_in[19];
    const float* v_b3  = (const float*)d_in[20];
    const float* a_w1  = (const float*)d_in[21];
    const float* a_bg  = (const float*)d_in[22];
    const float* a_bb  = (const float*)d_in[23];
    const float* a_bm  = (const float*)d_in[24];
    const float* a_bv  = (const float*)d_in[25];
    const float* a_w2  = (const float*)d_in[26];
    const float* a_b2  = (const float*)d_in[27];
    const float* g_w1  = (const float*)d_in[28];
    const float* g_bg  = (const float*)d_in[29];
    const float* g_bb  = (const float*)d_in[30];
    const float* g_bm  = (const float*)d_in[31];
    const float* g_bv  = (const float*)d_in[32];
    const float* g_w2  = (const float*)d_in[33];
    const float* g_b2  = (const float*)d_in[34];
    const float* ln_g  = (const float*)d_in[35];
    const float* ln_b  = (const float*)d_in[36];

    float* ws = (float*)d_ws;
    float* bufA = ws;                               // 32P : conv1 out, then excite
    float* bufB = ws + 32LL * PP;                   // 32P : qk, then t1/t2
    float* bufC = ws + 64LL * PP;                   // 36P : att, then gated (32P)
    float* v1   = ws + 100LL * PP;                  // 96*S
    float* v2   = v1 + 96 * SS;                     // 48*S
    float* v3   = v2 + 48 * SS;                     // 32*S
    __bf16* wf5  = (__bf16*)(v3 + 32 * SS);         // 25600 bf16 conv5x5 frags
    __bf16* wfpw = wf5 + 25600;                     // 1024 bf16 pointwise frags
    float* out = (float*)d_out;

    // qk path
    k_prep_w5<<<25, 64, 0, stream>>>(qk_w1, wf5);
    k_conv5x5<<<DD * HH * 2, 128, 0, stream>>>(Cost, (const v16bf*)wf5,
                                               qk_bg, qk_bb, qk_bm, qk_bv, bufA);
    k_prep_pw<<<1, 64, 0, stream>>>(qk_w2, wfpw);
    k_pointwise32<<<PP / 1024, 256, 0, stream>>>(bufA, (const v16bf*)wfpw,
        qk_b2, nullptr, nullptr, nullptr, nullptr, bufB, 1);
    // attention
    k_att<<<PP / 256, 256, 0, stream>>>(bufB, bufC);
    // to_v path
    k_conv3x3_v<<<96 * 32, 256, 0, stream>>>(Fmap, v_w1, v1_bg, v1_bb, v1_bm, v1_bv, v1);
    k_pw_small<<<(48 * SS) / 256, 256, 0, stream>>>(v1, v_w2, 96, 48,
        v2_bg, v2_bb, v2_bm, v2_bv, v2, 0);
    k_pw_small<<<(32 * SS) / 256, 256, 0, stream>>>(v2, v_w3, 48, 32,
        v_b3, nullptr, nullptr, nullptr, v3, 1);
    // excite + LayerNorm
    k_excite_ln<<<PP / 256, 256, 0, stream>>>(bufC, v3, ln_g, ln_b, bufA);
    // gate: t1 = lrelu(bn(a_w1 . excite)); gated = sigmoid(a_w2 . t1 + b) * Cost
    k_prep_pw<<<1, 64, 0, stream>>>(a_w1, wfpw);
    k_pointwise32<<<PP / 1024, 256, 0, stream>>>(bufA, (const v16bf*)wfpw,
        a_bg, a_bb, a_bm, a_bv, nullptr, bufB, 0);
    k_prep_pw<<<1, 64, 0, stream>>>(a_w2, wfpw);
    k_pointwise32<<<PP / 1024, 256, 0, stream>>>(bufB, (const v16bf*)wfpw,
        a_b2, nullptr, nullptr, nullptr, Cost, bufC, 2);
    // agg: out = g_w2 . lrelu(bn(g_w1 . gated)) + g_b2
    k_prep_pw<<<1, 64, 0, stream>>>(g_w1, wfpw);
    k_pointwise32<<<PP / 1024, 256, 0, stream>>>(bufC, (const v16bf*)wfpw,
        g_bg, g_bb, g_bm, g_bv, nullptr, bufB, 0);
    k_prep_pw<<<1, 64, 0, stream>>>(g_w2, wfpw);
    k_pointwise32<<<PP / 1024, 256, 0, stream>>>(bufB, (const v16bf*)wfpw,
        g_b2, nullptr, nullptr, nullptr, nullptr, out, 1);
}
